// ConditionalRandomField_87084756893974
// MI455X (gfx1250) — matrix operand order repaired and checked by reading, not compiled, not run on hardware
//
#include <hip/hip_runtime.h>
#include <stdint.h>

// CRF log-partition (forward algorithm) for gfx1250 / MI455X.
//
//   new[b,j] = pot[b,t,j] + m_b + log( sum_i exp(part[b,i]-m_b) * E[i,j] )
// E = exp(transitions), constant across steps, held in f16 B-fragment regs.
// Inner product: v_wmma_f32_16x16x32_f16 (wave32), K padded 52->64.
//
// 4 workgroups x 16 batches; 4 waves per workgroup, wave w owns tag columns
// [16w,16w+16). Per step: 2 barriers (partBuf double-buffered), potentials
// tile staged to LDS via global_load_async_to_lds_b32, mask consumed as one
// bit-packed word per step (branchless v_cndmask update). log() on the
// dependency chain uses raw v_log_f32 (safe: max-shift keeps acc in ~[0.5, 52·e]).

typedef __attribute__((ext_vector_type(16))) _Float16 v16h;
typedef __attribute__((ext_vector_type(8)))  float    v8f;

#define T2D        52      // NUM_TAGS + START + STOP
#define BSZ        64
#define SSZ        512
#define START_IDX  50
#define STOP_IDX   51
#define KP         64      // padded tag dimension (K and N)
#define NEG_BIG   (-1e30f)
#define LN2        0.6931471805599453f

__device__ __forceinline__ uint32_t lds_addr_of(const void* p) {
    // generic shared pointer: low 32 bits == LDS byte offset (aperture form)
    return (uint32_t)(uintptr_t)p;
}

__device__ __forceinline__ float fast_log(float x) {
    // v_log_f32 is log2; x is guaranteed normal and finite here.
    return __builtin_amdgcn_logf(x) * LN2;
}

__global__ __launch_bounds__(128)
void crf_forward_kernel(const float* __restrict__ pot,          // [B,S,T2]
                        const float* __restrict__ trans,        // [T2,T2]
                        const unsigned char* __restrict__ mask, // [B,S] bool
                        float* __restrict__ partial)            // [gridDim.x]
{
    __shared__ float    partBuf[2][16][KP + 1];  // double-buffered partition
    __shared__ float    potBuf[16][T2D];         // staged potentials tile
    __shared__ float    mrow[16];                // per-batch-row max
    __shared__ unsigned maskBits[SSZ];           // bit r = mask[b0+r][t]
    __shared__ float    rowVal[16];

    const int  tid     = threadIdx.x;
    const int  wave    = tid >> 5;           // 0..3 -> N tile
    const int  lane    = tid & 31;
    const int  halfSel = lane >> 4;          // 0 or 1
    const int  lrow    = lane & 15;
    const int  b0      = blockIdx.x * 16;    // batch block base
    const int  ncol    = wave * 16 + lrow;   // this lane's tag column
    const bool colOK   = (ncol < T2D);
    const int  jc      = colOK ? ncol : (T2D - 1);   // clamped (branchless OOB)

    // ---- bit-pack the mask once: maskBits[t] bit r = mask[b0+r][t] ----
    for (int t = tid; t < SSZ; t += 128) {
        unsigned w = 0;
        for (int r = 0; r < 16; ++r)
            w |= (unsigned)(mask[(size_t)(b0 + r) * SSZ + t] != 0) << r;
        maskBits[t] = w;
    }

    // ---- constant B fragments: E[i][j] = exp(trans[i][j]), f16, zero-padded ----
    // B, contiguous k order used here: lane<16 -> (K=kOff+e, N=lane); lane>=16 -> (K=kOff+16+e, N=lane-16)
    v16h bfrag[2];
    for (int kf = 0; kf < 2; ++kf) {
        const int kOff = kf * 32 + halfSel * 16;
        for (int e = 0; e < 16; ++e) {
            const int i = kOff + e;
            float ev = 0.0f;
            if (i < T2D && colOK) ev = __expf(trans[i * T2D + ncol]);
            bfrag[kf][e] = (_Float16)ev;
        }
    }

    // ---- part0 = pot[:,0,:] + trans[START,:]  (D-layout registers) ----
    float partReg[8];
    for (int r = 0; r < 8; ++r) {
        const int b = b0 + r + halfSel * 8;
        float v = NEG_BIG;
        if (colOK)
            v = pot[(size_t)b * SSZ * T2D + ncol] + trans[START_IDX * T2D + ncol];
        partReg[r] = v;
    }
    __syncthreads();   // maskBits ready

    // ---- sequential scan ----
    for (int t = 1; t < SSZ; ++t) {
        float (*pb)[KP + 1] = partBuf[t & 1];

        // publish current partition (padded cols stay -1e30 -> exp -> 0)
        for (int r = 0; r < 8; ++r)
            pb[r + halfSel * 8][ncol] = partReg[r];
        __syncthreads();   // B1

        // async-stage this step's 16x52 potentials tile into LDS (overlaps mrow)
        {
            const uint32_t ldsBase = lds_addr_of(&potBuf[0][0]);
            for (int idx = tid; idx < 16 * T2D; idx += 128) {
                const int row = idx / T2D;
                const int j   = idx - row * T2D;
                const uint64_t ga = (uint64_t)(uintptr_t)
                    &pot[((size_t)(b0 + row) * SSZ + t) * T2D + j];
                const uint32_t la = ldsBase + (uint32_t)idx * 4u;
                asm volatile("global_load_async_to_lds_b32 %0, %1, off"
                             :: "v"(la), "v"(ga) : "memory");
            }
        }
        if (tid < 16) {
            float m = NEG_BIG;
            for (int j = 0; j < T2D; ++j) m = fmaxf(m, pb[tid][j]);
            mrow[tid] = m;
        }
        if (t + 1 < SSZ)   // warm GL2 for next step's tile
            __builtin_prefetch(&pot[((size_t)(b0 + (tid & 15)) * SSZ + t + 1) * T2D], 0, 0);
        asm volatile("s_wait_asynccnt 0x0" ::: "memory");
        __syncthreads();   // B2

        // A fragments: exp(part - m), f16.
        // A, same contiguous k order as B: lane<16 -> M=lane,    K in {kOff+0..15}
        //                 lane>=16 -> M=lane-16, K in {kOff+16..31}
        const float m = mrow[lrow];
        v16h afrag[2];
        for (int kf = 0; kf < 2; ++kf) {
            const int kBase = kf * 32 + halfSel * 16;
            for (int e = 0; e < 16; ++e) {
                const int K = kBase + e;
                afrag[kf][e] = (_Float16)__expf(pb[lrow][K] - m);
            }
        }

        v8f acc = {};
        acc = __builtin_amdgcn_wmma_f32_16x16x32_f16(false, afrag[0], false, bfrag[0],
                                                     (short)0, acc, false, false);
        acc = __builtin_amdgcn_wmma_f32_16x16x32_f16(false, afrag[1], false, bfrag[1],
                                                     (short)0, acc, false, false);

        // branchless masked update: partReg = upd ? pot + m_b + log(acc) : partReg
        const unsigned bits = maskBits[t];   // uniform across workgroup
        for (int r = 0; r < 8; ++r) {
            const int   brow = r + halfSel * 8;
            const float nv   = potBuf[brow][jc] + mrow[brow] + fast_log(acc[r]);
            const bool  upd  = colOK && (((bits >> brow) & 1u) != 0u);
            partReg[r] = upd ? nv : partReg[r];
        }
        // no trailing barrier: partBuf is double-buffered; mrow/potBuf writers
        // of step t+1 only run after all waves pass B1(t+1).
    }

    // ---- final: logsumexp_i(part[b,i] + trans[i,STOP]), summed over batch ----
    for (int r = 0; r < 8; ++r)
        partBuf[0][r + halfSel * 8][ncol] = partReg[r];
    __syncthreads();

    if (tid < 16) {
        float m = NEG_BIG;
        for (int i = 0; i < T2D; ++i)
            m = fmaxf(m, partBuf[0][tid][i] + trans[i * T2D + STOP_IDX]);
        float s = 0.0f;
        for (int i = 0; i < T2D; ++i)
            s += __expf(partBuf[0][tid][i] + trans[i * T2D + STOP_IDX] - m);
        rowVal[tid] = m + fast_log(s);
    }
    __syncthreads();

    if (tid == 0) {
        float s = 0.0f;
        for (int r = 0; r < 16; ++r) s += rowVal[r];   // fixed order: deterministic
        partial[blockIdx.x] = s;
    }
}

__global__ void crf_finalize(const float* __restrict__ partial, float* __restrict__ out)
{
    float s = 0.0f;
    for (int i = 0; i < 4; ++i) s += partial[i];       // fixed order: deterministic
    out[0] = s;
}

extern "C" void kernel_launch(void* const* d_in, const int* in_sizes, int n_in,
                              void* d_out, int out_size, void* d_ws, size_t ws_size,
                              hipStream_t stream)
{
    const float*         pot   = (const float*)d_in[0];          // [64,512,52] f32
    const float*         trans = (const float*)d_in[1];          // [52,52] f32
    const unsigned char* mask  = (const unsigned char*)d_in[2];  // [64,512] bool
    float* out     = (float*)d_out;
    float* partial = (float*)d_ws;                               // 4 floats scratch

    crf_forward_kernel<<<4, 128, 0, stream>>>(pot, trans, mask, partial);
    crf_finalize<<<1, 1, 0, stream>>>(partial, out);
}
